// CausalSelfAttention_71829033058479
// MI455X (gfx1250) — matrix operand order repaired and checked by reading, not compiled, not run on hardware
//
#include <hip/hip_runtime.h>
#include <hip/hip_bf16.h>

// ---------------------------------------------------------------------------
// CDNA5 (gfx1250) fused CausalSelfAttention:
//   qkv = x @ W_attn^T          (bf16 WMMA, f32 accum)
//   flash attention (4 query sets share K/V)
//   out = comb @ W_proj^T       (bf16 WMMA, f32 out)
// Matmuls via v_wmma_f32_16x16x32_bf16. Tile staging: GEMM uses TDM
// tensor_load_to_lds (TENSORcnt, double-buffered); attention uses
// global_load_async_to_lds_b128 (ASYNCcnt) — both confirmed present.
// ---------------------------------------------------------------------------

typedef __attribute__((ext_vector_type(16))) __bf16        v16bf;
typedef __attribute__((ext_vector_type(8)))  float         v8f;
typedef __attribute__((ext_vector_type(4)))  unsigned int  v4u;
typedef __attribute__((ext_vector_type(8)))  int           v8i;
typedef __attribute__((ext_vector_type(4)))  int           v4i;

#ifndef __has_builtin
#define __has_builtin(x) 0
#endif

#if __has_builtin(__builtin_amdgcn_tensor_load_to_lds)
#define HAVE_TDM 1
#else
#define HAVE_TDM 0
#endif

#if __has_builtin(__builtin_amdgcn_global_load_async_to_lds_b128)
#define HAVE_ASYNC 1
#else
#define HAVE_ASYNC 0
#endif

#ifdef __HIP_DEVICE_COMPILE__
#if HAVE_TDM
#pragma message("CDNA5-PROBE: TDM tensor_load_to_lds path ENABLED (device)")
#else
#pragma message("CDNA5-PROBE: TDM builtin NOT FOUND (device)")
#endif
#if HAVE_ASYNC
#pragma message("CDNA5-PROBE: global_load_async_to_lds_b128 ENABLED (device)")
#else
#pragma message("CDNA5-PROBE: async-to-LDS builtin NOT FOUND (device)")
#endif
#endif

#if __has_builtin(__builtin_amdgcn_s_wait_tensorcnt)
#define WAIT_TENSORCNT(n) __builtin_amdgcn_s_wait_tensorcnt(n)
#else
#define WAIT_TENSORCNT(n) asm volatile("s_wait_tensorcnt %0" ::"n"(n))
#endif

#if __has_builtin(__builtin_amdgcn_s_wait_asynccnt)
#define WAIT_ASYNCCNT(n) __builtin_amdgcn_s_wait_asynccnt(n)
#else
#define WAIT_ASYNCCNT(n) asm volatile("s_wait_asynccnt %0" ::"n"(n))
#endif

// --- WMMA wrapper -----------------------------------------------------------
__device__ __forceinline__ v8f wmma_bf16(v16bf a, v16bf b, v8f c) {
  // (neg_a, A, neg_b, B, c_mod, C, reuse_a, reuse_b)
  return __builtin_amdgcn_wmma_f32_16x16x32_bf16(false, a, false, b,
                                                 (short)0, c, false, false);
}

// --- 16-element bf16 fragment load (two 16B chunks) -------------------------
typedef unsigned int u32x4 __attribute__((ext_vector_type(4)));
union FragU { v16bf v; u32x4 q[2]; };

template <typename P>
__device__ __forceinline__ v16bf load2x8(const P* p0, const P* p1) {
  FragU f;
  f.q[0] = *(const u32x4*)p0;
  f.q[1] = *(const u32x4*)p1;
  return f.v;
}

// --- Async global->LDS 16B per-lane copy ------------------------------------
#if HAVE_ASYNC
typedef __attribute__((address_space(1))) v4i as1_v4i;
typedef __attribute__((address_space(3))) v4i as3_v4i;
__device__ __forceinline__ void async_ld_b128(const void* g, void* l) {
  // param 0 per clang diagnostic: v4i* in AS1 (global), non-const.
  __builtin_amdgcn_global_load_async_to_lds_b128(
      (as1_v4i*)(v4i*)(void*)g, (as3_v4i*)(v4i*)l, 0, 0);
}
#endif

// --- Tensor Data Mover 2D tile load: global (row-major, bf16) -> LDS --------
#if HAVE_TDM
__device__ __forceinline__ void tdm_load_2d(unsigned lds_byte_off,
                                            const void* gptr,
                                            unsigned tile0,   // elems per row
                                            unsigned tile1,   // rows
                                            unsigned stride0  // row stride (elems)
                                            ) {
  unsigned long long ga = (unsigned long long)(__SIZE_TYPE__)gptr;
  v4u g0;
  g0[0] = 1u;                                            // count=1, user desc
  g0[1] = lds_byte_off;                                  // lds_addr [63:32]
  g0[2] = (unsigned)ga;                                  // global_addr lo
  g0[3] = (unsigned)((ga >> 32) & 0x01FFFFFFull) | (2u << 30); // addr hi | type=2
  v8i g1;
  g1[0] = (int)(1u << 16);                               // data_size=1 (2 bytes)
  g1[1] = (int)((tile0 & 0xFFFFu) << 16);                // tensor_dim0[15:0]
  g1[2] = (int)(((tile0 >> 16) & 0xFFFFu) |
                ((tile1 & 0xFFFFu) << 16));              // dim0 hi | dim1 lo
  g1[3] = (int)(((tile1 >> 16) & 0xFFFFu) |
                ((tile0 & 0xFFFFu) << 16));              // dim1 hi | tile_dim0
  g1[4] = (int)(tile1 & 0xFFFFu);                        // tile_dim1 (tile_dim2=0)
  g1[5] = (int)stride0;                                  // tensor_dim0_stride lo
  g1[6] = 0;
  g1[7] = 0;
  v4i z4 = {0, 0, 0, 0};
#if __clang_major__ >= 23
  v8i z8 = {0, 0, 0, 0, 0, 0, 0, 0};
  __builtin_amdgcn_tensor_load_to_lds(g0, g1, z4, z4, z8, 0);
#else
  __builtin_amdgcn_tensor_load_to_lds(g0, g1, z4, z4, 0);
#endif
}
#endif

// ---------------------------------------------------------------------------
// fp32 -> bf16 conversion (grid-stride)
// ---------------------------------------------------------------------------
__global__ __launch_bounds__(256) void cvt_f32_bf16(const float* __restrict__ in,
                                                    __bf16* __restrict__ out,
                                                    long n) {
  long i = (long)blockIdx.x * blockDim.x + threadIdx.x;
  long stride = (long)gridDim.x * blockDim.x;
  for (; i < n; i += stride) out[i] = (__bf16)in[i];
}

// ---------------------------------------------------------------------------
// GEMM: C[M,N] = A[M,K] * B[N,K]^T   (A,B bf16 row-major; OutT = __bf16|float)
// 128x128 block, 256 threads (8 waves), K-chunks of 32.
// Staging: TDM double-buffered > async single-buffer > plain.
// ---------------------------------------------------------------------------
template <typename OutT>
__global__ __launch_bounds__(256) void gemm_wmma(const __bf16* __restrict__ A,
                                                 const __bf16* __restrict__ B,
                                                 OutT* __restrict__ C,
                                                 int M, int N, int K) {
  const int tid  = threadIdx.x;
  const int wave = tid >> 5;
  const int lane = tid & 31;
  const int half = lane >> 4;
  const int l16  = lane & 15;
  const int m0   = blockIdx.y * 128;
  const int n0   = blockIdx.x * 128;

  // [A buf0][A buf1][B buf0][B buf1], each 128x32 bf16 = 4096 elems (8 KB)
  __shared__ __bf16 smem[4 * 4096];

  v8f acc[8];
#pragma unroll
  for (int j = 0; j < 8; ++j) acc[j] = v8f{};

  const int KC = K >> 5;

#if HAVE_TDM
  if (wave == 0) {
    tdm_load_2d(0u,              A + (__SIZE_TYPE__)m0 * K, 32u, 128u, (unsigned)K);
    tdm_load_2d(2u * 4096u * 2u, B + (__SIZE_TYPE__)n0 * K, 32u, 128u, (unsigned)K);
  }
#endif

  for (int kc = 0; kc < KC; ++kc) {
#if HAVE_TDM
    const int p = kc & 1;
    if (wave == 0) {
      if (kc + 1 < KC) {
        tdm_load_2d((unsigned)((p ^ 1) * 4096 * 2),
                    A + (__SIZE_TYPE__)m0 * K + (kc + 1) * 32, 32u, 128u, (unsigned)K);
        tdm_load_2d((unsigned)((2 + (p ^ 1)) * 4096 * 2),
                    B + (__SIZE_TYPE__)n0 * K + (kc + 1) * 32, 32u, 128u, (unsigned)K);
        WAIT_TENSORCNT(2);
      } else {
        WAIT_TENSORCNT(0);
      }
    }
    __syncthreads();
    const __bf16* as = smem + p * 4096;
    const __bf16* bs = smem + (2 + p) * 4096;
#elif HAVE_ASYNC
    __syncthreads();  // protect previous iteration's reads
    // 128x32 bf16 tile = 512 16B chunks; 256 threads x 2 chunks each.
#pragma unroll
    for (int u = 0; u < 2; ++u) {
      const int i = tid + u * 256;
      const int r = i >> 2, c = i & 3;  // 4 chunks of 8 elems per 32-elem row
      async_ld_b128(A + (__SIZE_TYPE__)(m0 + r) * K + kc * 32 + c * 8,
                    smem + r * 32 + c * 8);
      async_ld_b128(B + (__SIZE_TYPE__)(n0 + r) * K + kc * 32 + c * 8,
                    smem + 4096 + r * 32 + c * 8);
    }
    WAIT_ASYNCCNT(0);
    __syncthreads();
    const __bf16* as = smem;
    const __bf16* bs = smem + 4096;
#else
    __syncthreads();
    for (int i = tid; i < 4096; i += 256) {
      int r = i >> 5, c = i & 31;
      smem[i]        = A[(__SIZE_TYPE__)(m0 + r) * K + kc * 32 + c];
      smem[4096 + i] = B[(__SIZE_TYPE__)(n0 + r) * K + kc * 32 + c];
    }
    __syncthreads();
    const __bf16* as = smem;
    const __bf16* bs = smem + 4096;
#endif

    // A fragment: M = l16 (+wave*16), K = half*16 + e (contiguous 16)
    const __bf16* arow = as + (wave * 16 + l16) * 32;
    v16bf afrag = load2x8(arow + half * 16, arow + half * 16 + 8);

#pragma unroll
    for (int j = 0; j < 8; ++j) {
      // B fragment: N = j*16 + l16, K = half*16 + e (contiguous 16)
      const __bf16* brow = bs + (j * 16 + l16) * 32 + half * 16;
      v16bf bfrag = load2x8(brow, brow + 8);
      acc[j] = wmma_bf16(afrag, bfrag, acc[j]);
    }

#if HAVE_TDM
    __syncthreads();  // protect buffers before next-iteration TDM overwrite
#endif
  }

  // C/D layout: M = (lane/16)*8 + r, N = lane%16
#pragma unroll
  for (int j = 0; j < 8; ++j) {
#pragma unroll
    for (int r = 0; r < 8; ++r) {
      __SIZE_TYPE__ row = (__SIZE_TYPE__)(m0 + wave * 16 + half * 8 + r);
      __SIZE_TYPE__ col = (__SIZE_TYPE__)(n0 + j * 16 + l16);
      C[row * N + col] = (OutT)acc[j][r];
    }
  }
}

// ---------------------------------------------------------------------------
// Flash attention, bf16 WMMA, f32 online softmax.
// qkv layout: [B*T, 6144] = [q0|q1|q2|q3|K|V] each 1024 wide, head h at h*64.
// grid: 4096 blocks = (b:8, h:16, g:4, qblk:8); 256 threads (8 waves),
// each wave owns 16 query rows; kv tiles of 32.
// Staging: async-to-LDS > TDM > plain.
// ---------------------------------------------------------------------------
__global__ __launch_bounds__(256) void attn_wmma(const __bf16* __restrict__ qkv,
                                                 __bf16* __restrict__ comb) {
  constexpr int C = 1024, HD = 64, Tt = 1024, QW = 6144, GW = 4096;
  const int bid  = blockIdx.x;
  const int qblk = bid & 7;
  const int g    = (bid >> 3) & 3;
  const int h    = (bid >> 5) & 15;
  const int b    = bid >> 9;
  const int q0   = qblk * 128;

  const int tid  = threadIdx.x;
  const int wave = tid >> 5;
  const int lane = tid & 31;
  const int half = lane >> 4;
  const int l16  = lane & 15;

  // Ks[32][64] | Vs[32][64] | P: 8 waves x 16x32
  __shared__ __bf16 smem[8192];
  const __bf16* Ks = smem;
  const __bf16* Vs = smem + 2048;
  __bf16* Pm = smem + 4096 + wave * 512;

  // Q fragments (A layout), hd split into two K=32 chunks
  const __bf16* qbase =
      qkv + (__SIZE_TYPE__)(b * Tt + q0 + wave * 16 + l16) * QW + g * C + h * HD;
  v16bf qa0 = load2x8(qbase + half * 16, qbase + half * 16 + 8);
  v16bf qa1 = load2x8(qbase + 32 + half * 16, qbase + 40 + half * 16);

  v8f o[4] = {};
  float mrow[8], lrow[8];
#pragma unroll
  for (int r = 0; r < 8; ++r) { mrow[r] = -3.0e38f; lrow[r] = 0.0f; }

  const float scale = 0.125f;   // 1/sqrt(64)
  const int nkv = q0 / 32 + 4;  // kv tiles of 32 covering [0, q0+128)

  for (int kt = 0; kt < nkv; ++kt) {
    const int kb = kt * 32;
#if HAVE_ASYNC
    // 32x64 bf16 tile = 4 KB = 256 lanes x 16B : one async b128 per thread/tile
    {
      const int r = tid >> 3, c = tid & 7;  // 8 chunks of 8 elems per 64-elem row
      const __SIZE_TYPE__ grow = (__SIZE_TYPE__)(b * Tt + kb + r) * QW + h * HD + c * 8;
      async_ld_b128(qkv + grow + 4 * C, smem + r * 64 + c * 8);         // K
      async_ld_b128(qkv + grow + 5 * C, smem + 2048 + r * 64 + c * 8);  // V
    }
    WAIT_ASYNCCNT(0);
    __syncthreads();
#elif HAVE_TDM
    if (wave == 0) {
      tdm_load_2d(0u,
                  qkv + (__SIZE_TYPE__)(b * Tt + kb) * QW + 4 * C + h * HD,
                  64u, 32u, (unsigned)QW);
      tdm_load_2d(4096u,
                  qkv + (__SIZE_TYPE__)(b * Tt + kb) * QW + 5 * C + h * HD,
                  64u, 32u, (unsigned)QW);
      WAIT_TENSORCNT(0);
    }
    __syncthreads();
#else
    __syncthreads();
    for (int i = tid; i < 2048; i += 256) {
      int r = i >> 6, c2 = i & 63;
      smem[i]        = qkv[(__SIZE_TYPE__)(b * Tt + kb + r) * QW + 4 * C + h * HD + c2];
      smem[2048 + i] = qkv[(__SIZE_TYPE__)(b * Tt + kb + r) * QW + 5 * C + h * HD + c2];
    }
    __syncthreads();
#endif

    // S = scale * Q K^T, two 16x16 tiles (kv cols 0-15, 16-31)
    float sv[2][8];
#pragma unroll
    for (int t = 0; t < 2; ++t) {
      v8f s = {};
#pragma unroll
      for (int c = 0; c < 2; ++c) {
        // B frag: n = t*16+l16 (kv row in Ks), k = c*32 + half*16 + e (hd)
        const __bf16* krow = Ks + (t * 16 + l16) * 64 + c * 32 + half * 16;
        v16bf kf = load2x8(krow, krow + 8);
        s = wmma_bf16(c == 0 ? qa0 : qa1, kf, s);
      }
#pragma unroll
      for (int r = 0; r < 8; ++r) {
        const int colg = kb + t * 16 + l16;
        const int rowg = q0 + wave * 16 + half * 8 + r;
        sv[t][r] = (colg <= rowg) ? s[r] * scale : -1.0e30f;
      }
    }

    // online softmax update (row = 16 lanes of same half-wave)
#pragma unroll
    for (int r = 0; r < 8; ++r) {
      float sm = fmaxf(sv[0][r], sv[1][r]);
      sm = fmaxf(sm, __shfl_xor(sm, 1, 32));
      sm = fmaxf(sm, __shfl_xor(sm, 2, 32));
      sm = fmaxf(sm, __shfl_xor(sm, 4, 32));
      sm = fmaxf(sm, __shfl_xor(sm, 8, 32));
      const float mn  = fmaxf(mrow[r], sm);
      const float fac = __expf(mrow[r] - mn);
      mrow[r] = mn;
      sv[0][r] = __expf(sv[0][r] - mn);
      sv[1][r] = __expf(sv[1][r] - mn);
      float rs = sv[0][r] + sv[1][r];
      rs += __shfl_xor(rs, 1, 32);
      rs += __shfl_xor(rs, 2, 32);
      rs += __shfl_xor(rs, 4, 32);
      rs += __shfl_xor(rs, 8, 32);
      lrow[r] = lrow[r] * fac + rs;
      o[0][r] *= fac; o[1][r] *= fac; o[2][r] *= fac; o[3][r] *= fac;
    }

    // stage P (D layout -> LDS row-major 16x32) for use as next A fragment
#pragma unroll
    for (int t = 0; t < 2; ++t)
#pragma unroll
      for (int r = 0; r < 8; ++r)
        Pm[(half * 8 + r) * 32 + t * 16 + l16] = (__bf16)sv[t][r];
    asm volatile("s_wait_dscnt 0" ::: "memory");

    // O += P @ V : A = P(16x32), B = V slice (32 x 16) per hd chunk
    const __bf16* prow = Pm + l16 * 32;
    v16bf pa = load2x8(prow + half * 16, prow + half * 16 + 8);
#pragma unroll
    for (int j = 0; j < 4; ++j) {
      v16bf vf;
#pragma unroll
      for (int e = 0; e < 16; ++e)
        vf[e] = Vs[(half * 16 + e) * 64 + j * 16 + l16];
      o[j] = wmma_bf16(pa, vf, o[j]);
    }
    __syncthreads();  // before Ks/Vs are overwritten
  }

  // epilogue: O / l -> comb[b,t, g*1024 + h*64 + d] (bf16)
#pragma unroll
  for (int j = 0; j < 4; ++j) {
#pragma unroll
    for (int r = 0; r < 8; ++r) {
      const int rowg = q0 + wave * 16 + half * 8 + r;
      const float val = o[j][r] / lrow[r];
      comb[(__SIZE_TYPE__)(b * Tt + rowg) * GW + g * C + h * HD + j * 16 + l16] =
          (__bf16)val;
    }
  }
}

// ---------------------------------------------------------------------------
// Host launcher
// ---------------------------------------------------------------------------
extern "C" void kernel_launch(void* const* d_in, const int* in_sizes, int n_in,
                              void* d_out, int out_size, void* d_ws, size_t ws_size,
                              hipStream_t stream) {
  const float* x  = (const float*)d_in[0];   // [8,1024,1024]
  const float* Wa = (const float*)d_in[1];   // [6144,1024]
  const float* Wp = (const float*)d_in[2];   // [1024,4096]
  float* out      = (float*)d_out;           // [8,1024,1024]

  const long NX  = 8L * 1024 * 1024;   // 8388608
  const long NWA = 6144L * 1024;       // 6291456
  const long NWP = 1024L * 4096;       // 4194304

  char* ws = (char*)d_ws;
  __bf16* xb   = (__bf16*)(ws);
  __bf16* Wab  = (__bf16*)(ws + 16777216);
  __bf16* Wpb  = (__bf16*)(ws + 16777216 + 12582912);
  __bf16* qkv  = (__bf16*)(ws + 16777216 + 12582912 + 8388608);
  __bf16* comb = (__bf16*)(ws + 16777216 + 12582912 + 8388608 + 100663296);

  cvt_f32_bf16<<<4096, 256, 0, stream>>>(x, xb, NX);
  cvt_f32_bf16<<<4096, 256, 0, stream>>>(Wa, Wab, NWA);
  cvt_f32_bf16<<<4096, 256, 0, stream>>>(Wp, Wpb, NWP);

  // qkv[8192,6144] = xb[8192,1024] @ Wab[6144,1024]^T
  gemm_wmma<__bf16><<<dim3(48, 64), 256, 0, stream>>>(xb, Wab, qkv, 8192, 6144, 1024);

  // attention -> comb[8192,4096] (bf16)
  attn_wmma<<<4096, 256, 0, stream>>>(qkv, comb);

  // out[8192,1024] = comb[8192,4096] @ Wpb[1024,4096]^T  (f32 out)
  gemm_wmma<float><<<dim3(8, 64), 256, 0, stream>>>(comb, Wpb, out, 8192, 1024, 4096);
}
